// MoELayer_42159398977858
// MI455X (gfx1250) — matrix operand order repaired and checked by reading, not compiled, not run on hardware
//
#include <hip/hip_runtime.h>
#include <math.h>

// ---------------- problem constants ----------------
#define N_TOK 8192     // B*T
#define D_DIM 512
#define H_DIM 1024
#define E_NUM 8
#define K_TOP 2

#define TM    64       // tokens per expert tile
#define HB    256      // H columns per GEMM1 chunk
#define NB    128      // D columns per GEMM2 chunk
#define HPAD  1032     // padded LDS row stride (bf16 elems), 16B-aligned rows
#define SK    128      // K elements staged per async double-buffer stage

// LDS layout (bytes) inside the expert kernel
#define LDS_HT_BYTES   ((size_t)TM * HPAD * 2)          // 132096
#define LDS_TOK_BYTES  (TM * 8)                         // 512 (tok + wgt)
#define LDS_BSTAGE_OFF (LDS_HT_BYTES + LDS_TOK_BYTES)   // 132608 (16B aligned)
#define BUF_BYTES      (HB * SK * 2)                    // 65536 per buffer
#define BUF_USH        (HB * SK)                        // 32768 ushorts

// ---------------- WMMA vector types ----------------
typedef __attribute__((ext_vector_type(4)))  unsigned int v4u;
typedef __attribute__((ext_vector_type(8)))  float        v8f;
typedef __attribute__((ext_vector_type(16))) __bf16       v16bf;

union Frag { v16bf m; v4u q[2]; };

__device__ __forceinline__ unsigned short f2bf(float f) {
    unsigned int u = __float_as_uint(f);
    unsigned int r = u + 0x7FFFu + ((u >> 16) & 1u);   // round-to-nearest-even
    return (unsigned short)(r >> 16);
}

// async global -> LDS 16B DMA (ASYNCcnt-tracked); vdst = LDS byte address VGPR
__device__ __forceinline__ void async_ldscpy_b128(unsigned int ldsByteAddr,
                                                  const unsigned short* gsrc) {
    asm volatile("global_load_async_to_lds_b128 %0, %1, off"
                 :: "v"(ldsByteAddr), "v"(gsrc) : "memory");
}
__device__ __forceinline__ void wait_async_le16() { asm volatile("s_wait_asynccnt 16" ::: "memory"); }
__device__ __forceinline__ void wait_async_le8()  { asm volatile("s_wait_asynccnt 8"  ::: "memory"); }
__device__ __forceinline__ void wait_async_0()    { asm volatile("s_wait_asynccnt 0"  ::: "memory"); }

// ---------------- workspace layout (bytes) ----------------
#define WS_XBF   0
#define WS_W1T   (WS_XBF + (size_t)N_TOK * D_DIM * 2)
#define WS_W2T   (WS_W1T + (size_t)E_NUM * H_DIM * D_DIM * 2)
#define WS_CNT   (WS_W2T + (size_t)E_NUM * D_DIM * H_DIM * 2)
#define WS_TOK   (WS_CNT + 64)
#define WS_WGT   (WS_TOK + (size_t)E_NUM * N_TOK * 4)

// ---------------------------------------------------------------------------
// 0) zero the routed-output region + expert counters
// ---------------------------------------------------------------------------
__global__ void moe_zero_kernel(float* __restrict__ out, int* __restrict__ counts) {
    long long i = (long long)blockIdx.x * blockDim.x + threadIdx.x;
    if (i < (long long)N_TOK * D_DIM) out[i] = 0.0f;
    if (i < E_NUM) counts[i] = 0;
}

// ---------------------------------------------------------------------------
// 1) router: logits, top-2 softmax, aux outputs, expert lists, x -> bf16
// ---------------------------------------------------------------------------
__global__ __launch_bounds__(256) void moe_router_kernel(
    const float* __restrict__ x, const float* __restrict__ Wg,
    float* __restrict__ aux_logits, int* __restrict__ aux_idx,
    float* __restrict__ aux_w,
    unsigned short* __restrict__ xbf,
    int* __restrict__ counts, int* __restrict__ toklist,
    float* __restrict__ wgtlist)
{
    __shared__ float wg_s[D_DIM * E_NUM];                 // 16 KB
    for (int i = threadIdx.x; i < D_DIM * E_NUM; i += blockDim.x) wg_s[i] = Wg[i];
    __syncthreads();

    int n = blockIdx.x * blockDim.x + threadIdx.x;
    if (n >= N_TOK) return;

    const float* xr = x + (size_t)n * D_DIM;
    float acc[E_NUM];
#pragma unroll
    for (int e = 0; e < E_NUM; ++e) acc[e] = 0.0f;

    for (int d = 0; d < D_DIM; ++d) {
        float xv = xr[d];
        xbf[(size_t)n * D_DIM + d] = f2bf(xv);
#pragma unroll
        for (int e = 0; e < E_NUM; ++e) acc[e] += xv * wg_s[d * E_NUM + e];
    }

#pragma unroll
    for (int e = 0; e < E_NUM; ++e) aux_logits[(size_t)n * E_NUM + e] = acc[e];

    int i1 = 0;
#pragma unroll
    for (int e = 1; e < E_NUM; ++e) if (acc[e] > acc[i1]) i1 = e;
    int i2 = (i1 == 0) ? 1 : 0;
#pragma unroll
    for (int e = 0; e < E_NUM; ++e) if (e != i1 && acc[e] > acc[i2]) i2 = e;

    float l1 = acc[i1], l2 = acc[i2];
    float t  = expf(l2 - l1);                 // <= 1, stable
    float w1v = 1.0f / (1.0f + t);
    float w2v = t * w1v;

    aux_idx[n * K_TOP + 0] = i1;  aux_idx[n * K_TOP + 1] = i2;
    aux_w[n * K_TOP + 0]  = w1v;  aux_w[n * K_TOP + 1]  = w2v;

    int p1 = atomicAdd(&counts[i1], 1);
    toklist[(size_t)i1 * N_TOK + p1] = n;  wgtlist[(size_t)i1 * N_TOK + p1] = w1v;
    int p2 = atomicAdd(&counts[i2], 1);
    toklist[(size_t)i2 * N_TOK + p2] = n;  wgtlist[(size_t)i2 * N_TOK + p2] = w2v;
}

// ---------------------------------------------------------------------------
// 2) transpose+convert: in [E][R][C] f32 -> out [E][C][R] bf16
// ---------------------------------------------------------------------------
__global__ void moe_transpose_bf16_kernel(const float* __restrict__ in,
                                          unsigned short* __restrict__ out,
                                          int R, int C)
{
    long long id = (long long)blockIdx.x * blockDim.x + threadIdx.x;
    long long total = (long long)E_NUM * R * C;
    if (id >= total) return;
    int e = (int)(id / ((long long)R * C));
    int rem = (int)(id - (long long)e * R * C);
    int r = rem / C, c = rem % C;
    out[((size_t)e * C + c) * R + r] = f2bf(in[id]);
}

// ---------------------------------------------------------------------------
// 3) fused expert FFN: gather 64 tokens, GEMM1+GELU (h in LDS), GEMM2+combine
//    B operands double-buffered in LDS via async global->LDS DMA.
// ---------------------------------------------------------------------------
__global__ __launch_bounds__(256) void moe_expert_ffn_kernel(
    const unsigned short* __restrict__ xbf,   // [N][D] bf16
    const unsigned short* __restrict__ w1t,   // [E][H][D] bf16
    const unsigned short* __restrict__ w2t,   // [E][D][H] bf16
    const float* __restrict__ b1,             // [E][H]
    const float* __restrict__ b2,             // [E][D]
    const int*   __restrict__ counts,
    const int*   __restrict__ toklist,        // [E][N]
    const float* __restrict__ wgtlist,        // [E][N]
    float* __restrict__ out)                  // [N][D]
{
    extern __shared__ char smem[];
    unsigned short* hTile  = (unsigned short*)smem;                  // [TM][HPAD]
    int*   tokS   = (int*)(smem + LDS_HT_BYTES);
    float* wgtS   = (float*)(tokS + TM);
    unsigned short* bStage = (unsigned short*)(smem + LDS_BSTAGE_OFF); // 2 x 64KB
    // LDS byte address of bStage (generic shared ptr low 32 bits = LDS offset)
    const unsigned int bStageLds = (unsigned int)(size_t)(void*)bStage;

    const int e        = blockIdx.y;
    const int cnt      = counts[e];
    const int tileBase = blockIdx.x * TM;
    if (tileBase >= cnt) return;                              // uniform exit

    const int tid = threadIdx.x;
    if (tid < TM) {
        int slot = tileBase + tid;
        int tok = 0; float w = 0.0f;
        if (slot < cnt) {
            tok = toklist[(size_t)e * N_TOK + slot];
            w   = wgtlist[(size_t)e * N_TOK + slot];
        }
        tokS[tid] = tok; wgtS[tid] = w;
    }
    __syncthreads();

    const int wave    = tid >> 5;
    const int lane    = tid & 31;
    const int rowTile = wave & 3;        // 4 row tiles of 16
    const int colGrp  = wave >> 2;       // 2 column groups
    const int rowBase = rowTile * 16;
    const int l15     = lane & 15;
    const int hi      = lane >> 4;       // lane half
    const int koffA   = hi * 8;          // A: K 0-7/16-23 vs 8-15/24-31
    const int koffB   = hi * 8;          // B: K 0-7/16-23 vs 8-15/24-31

    const int tokA = tokS[rowBase + l15];
    const unsigned short* aRow = xbf + (size_t)tokA * D_DIM;

    // ---------------- GEMM1: h = GELU(x @ W1 + b1) -> LDS (bf16) -----------
    const unsigned short* w1e = w1t + (size_t)e * H_DIM * D_DIM;
    for (int hb = 0; hb < H_DIM; hb += HB) {
        v8f acc[8];
#pragma unroll
        for (int j = 0; j < 8; ++j) acc[j] = (v8f){0,0,0,0,0,0,0,0};

        // prologue: stage 0 -> buffer 0  (256 cols x SK, 16 x 16B per thread)
#pragma unroll
        for (int c = 0; c < 16; ++c) {
            int i = c * 256 + tid;
            int col = i >> 4, part = i & 15;
            async_ldscpy_b128(bStageLds + (unsigned)(col * SK + part * 8) * 2,
                              w1e + (size_t)(hb + col) * D_DIM + part * 8);
        }

        for (int s = 0; s < D_DIM / SK; ++s) {               // 4 stages
            if (s + 1 < D_DIM / SK) {
                unsigned int dstB = bStageLds + ((s + 1) & 1) * BUF_BYTES;
                int kkS = (s + 1) * SK;
#pragma unroll
                for (int c = 0; c < 16; ++c) {
                    int i = c * 256 + tid;
                    int col = i >> 4, part = i & 15;
                    async_ldscpy_b128(dstB + (unsigned)(col * SK + part * 8) * 2,
                                      w1e + (size_t)(hb + col) * D_DIM + kkS + part * 8);
                }
                wait_async_le16();          // in-order => stage s resident
            } else {
                wait_async_0();
            }
            __syncthreads();

            const unsigned short* bufB = bStage + (s & 1) * BUF_USH;
#pragma unroll
            for (int kk2 = 0; kk2 < SK; kk2 += 32) {
                int kk = s * SK + kk2;
                Frag A;
                A.q[0] = *reinterpret_cast<const v4u*>(aRow + kk + koffA);
                A.q[1] = *reinterpret_cast<const v4u*>(aRow + kk + koffA + 16);
                __builtin_prefetch(aRow + kk + 32, 0, 3);    // near prefetch
#pragma unroll
                for (int j = 0; j < 8; ++j) {
                    int colL = (colGrp * 8 + j) * 16 + l15;  // 0..255
                    const unsigned short* bp = bufB + colL * SK + kk2 + koffB;
                    Frag Bf;
                    Bf.q[0] = *reinterpret_cast<const v4u*>(bp);     // ds_load_b128
                    Bf.q[1] = *reinterpret_cast<const v4u*>(bp + 16);
                    acc[j] = __builtin_amdgcn_wmma_f32_16x16x32_bf16(
                        false, A.m, false, Bf.m, (short)0, acc[j], false, false);
                }
            }
            __syncthreads();                                 // safe to overwrite
        }

        // bias + exact GELU, store bf16 into LDS h tile
#pragma unroll
        for (int j = 0; j < 8; ++j) {
            int col  = hb + (colGrp * 8 + j) * 16 + l15;
            float bs = b1[(size_t)e * H_DIM + col];
#pragma unroll
            for (int r = 0; r < 8; ++r) {
                int row = rowBase + r + hi * 8;
                float v = acc[j][r] + bs;
                float g = 0.5f * v * (1.0f + erff(v * 0.70710678118f));
                hTile[row * HPAD + col] = f2bf(g);
            }
        }
    }
    __syncthreads();

    // ---------------- GEMM2: out += w * (h @ W2 + b2) ----------------------
    const unsigned short* w2e  = w2t + (size_t)e * D_DIM * H_DIM;
    const unsigned short* aLds = hTile + (rowBase + l15) * HPAD;
    for (int nb = 0; nb < D_DIM; nb += NB) {
        v8f acc[4];
#pragma unroll
        for (int j = 0; j < 4; ++j) acc[j] = (v8f){0,0,0,0,0,0,0,0};

        // prologue: stage 0 -> buffer 0  (128 cols x SK, 8 x 16B per thread)
#pragma unroll
        for (int c = 0; c < 8; ++c) {
            int i = c * 256 + tid;
            int col = i >> 4, part = i & 15;
            async_ldscpy_b128(bStageLds + (unsigned)(col * SK + part * 8) * 2,
                              w2e + (size_t)(nb + col) * H_DIM + part * 8);
        }

        for (int s = 0; s < H_DIM / SK; ++s) {               // 8 stages
            if (s + 1 < H_DIM / SK) {
                unsigned int dstB = bStageLds + ((s + 1) & 1) * BUF_BYTES;
                int kkS = (s + 1) * SK;
#pragma unroll
                for (int c = 0; c < 8; ++c) {
                    int i = c * 256 + tid;
                    int col = i >> 4, part = i & 15;
                    async_ldscpy_b128(dstB + (unsigned)(col * SK + part * 8) * 2,
                                      w2e + (size_t)(nb + col) * H_DIM + kkS + part * 8);
                }
                wait_async_le8();
            } else {
                wait_async_0();
            }
            __syncthreads();

            const unsigned short* bufB = bStage + (s & 1) * BUF_USH;
#pragma unroll
            for (int kk2 = 0; kk2 < SK; kk2 += 32) {
                int kk = s * SK + kk2;
                Frag A;
                A.q[0] = *reinterpret_cast<const v4u*>(aLds + kk + koffA);      // ds_load_b128
                A.q[1] = *reinterpret_cast<const v4u*>(aLds + kk + koffA + 16);
#pragma unroll
                for (int j = 0; j < 4; ++j) {
                    int colL = (colGrp * 4 + j) * 16 + l15;  // 0..127
                    const unsigned short* bp = bufB + colL * SK + kk2 + koffB;
                    Frag Bf;
                    Bf.q[0] = *reinterpret_cast<const v4u*>(bp);
                    Bf.q[1] = *reinterpret_cast<const v4u*>(bp + 16);
                    acc[j] = __builtin_amdgcn_wmma_f32_16x16x32_bf16(
                        false, A.m, false, Bf.m, (short)0, acc[j], false, false);
                }
            }
            __syncthreads();
        }

#pragma unroll
        for (int j = 0; j < 4; ++j) {
            int col  = nb + (colGrp * 4 + j) * 16 + l15;
            float bs = b2[(size_t)e * D_DIM + col];
#pragma unroll
            for (int r = 0; r < 8; ++r) {
                int row  = rowBase + r + hi * 8;
                int slot = tileBase + row;
                if (slot < cnt) {
                    float val = (acc[j][r] + bs) * wgtS[row];
                    atomicAdd(&out[(size_t)tokS[row] * D_DIM + col], val);
                }
            }
        }
    }
}

// ---------------------------------------------------------------------------
// launch
// ---------------------------------------------------------------------------
extern "C" void kernel_launch(void* const* d_in, const int* in_sizes, int n_in,
                              void* d_out, int out_size, void* d_ws, size_t ws_size,
                              hipStream_t stream) {
    const float* x  = (const float*)d_in[0];   // [B,T,D]
    const float* Wg = (const float*)d_in[1];   // [D,E]
    const float* W1 = (const float*)d_in[2];   // [E,D,H]
    const float* b1 = (const float*)d_in[3];   // [E,H]
    const float* W2 = (const float*)d_in[4];   // [E,H,D]
    const float* b2 = (const float*)d_in[5];   // [E,D]

    float* outp       = (float*)d_out;                                  // [N,D]
    float* aux_logits = outp + (size_t)N_TOK * D_DIM;                   // [N,E]
    int*   aux_idx    = (int*)(aux_logits + (size_t)N_TOK * E_NUM);     // [N,K]
    float* aux_w      = (float*)(aux_idx + (size_t)N_TOK * K_TOP);      // [N,K]

    char* ws = (char*)d_ws;
    unsigned short* xbf = (unsigned short*)(ws + WS_XBF);
    unsigned short* w1t = (unsigned short*)(ws + WS_W1T);
    unsigned short* w2t = (unsigned short*)(ws + WS_W2T);
    int*   counts  = (int*)(ws + WS_CNT);
    int*   toklist = (int*)(ws + WS_TOK);
    float* wgtlist = (float*)(ws + WS_WGT);

    // 0) zero out + counters
    {
        long long total = (long long)N_TOK * D_DIM;
        int blocks = (int)((total + 255) / 256);
        moe_zero_kernel<<<blocks, 256, 0, stream>>>(outp, counts);
    }
    // 1) router (+ x -> bf16)
    moe_router_kernel<<<N_TOK / 256, 256, 0, stream>>>(
        x, Wg, aux_logits, aux_idx, aux_w, xbf, counts, toklist, wgtlist);
    // 2) W1 [E][D][H] -> [E][H][D] bf16 ; W2 [E][H][D] -> [E][D][H] bf16
    {
        long long total = (long long)E_NUM * D_DIM * H_DIM;
        int blocks = (int)((total + 255) / 256);
        moe_transpose_bf16_kernel<<<blocks, 256, 0, stream>>>(W1, w1t, D_DIM, H_DIM);
        moe_transpose_bf16_kernel<<<blocks, 256, 0, stream>>>(W2, w2t, H_DIM, D_DIM);
    }
    // 3) fused expert FFN
    {
        dim3 grid((N_TOK + TM - 1) / TM, E_NUM);   // 128 x 8, early-exit on count
        size_t shmem = LDS_BSTAGE_OFF + 2 * BUF_BYTES;   // 263,680 B of WGP LDS
        moe_expert_ffn_kernel<<<grid, dim3(256), shmem, stream>>>(
            xbf, w1t, w2t, b1, b2, counts, toklist, wgtlist, outp);
    }
}
